// RayMarcher_2439541424393
// MI455X (gfx1250) — matrix operand order repaired and checked by reading, not compiled, not run on hardware
//
#include <hip/hip_runtime.h>

typedef __attribute__((ext_vector_type(16))) _Float16 v16h;
typedef __attribute__((ext_vector_type(8)))  _Float16 v8h;
typedef __attribute__((ext_vector_type(8)))  float    v8f;

#define NRAYS 16384
#define SSTEPS 128

#if defined(__has_builtin)
#if __has_builtin(__builtin_amdgcn_wmma_f16_16x16x32_f16)
#define HAVE_WMMA_F16D 1
#endif
#endif

// d_ws layout (bytes):
//     0 : W1t  f16 [64 col][64 K]   (8192)
//  8192 : W2t  f16 [64 col][64 K]   (8192)
// 16384 : WeffT f16 [16 col][64 K]  (2048)   col0=sigma, col1..3=fused rgb, 4..15=0
// 18432 : b_eff f32 [16]            (64)
// 18496 : dt    f32                 (4)

__global__ void prep_kernel(const float* __restrict__ W1, const float* __restrict__ W2,
                            const float* __restrict__ Wout, const float* __restrict__ bout,
                            const float* __restrict__ Wrgb, const float* __restrict__ brgb,
                            const float* __restrict__ tmin, const float* __restrict__ tmax,
                            void* ws) {
  _Float16* w1t  = (_Float16*)ws;
  _Float16* w2t  = (_Float16*)((char*)ws + 8192);
  _Float16* weff = (_Float16*)((char*)ws + 16384);
  float*    beff = (float*)((char*)ws + 18432);
  float*    dtp  = (float*)((char*)ws + 18496);
  int tid = threadIdx.x;
  for (int i = tid; i < 64*64; i += blockDim.x) {
    int c = i >> 6, k = i & 63;
    w1t[i] = (_Float16)((k < 63) ? W1[k*64 + c] : 0.f);   // W1 is 63x64; pad K row 63 with 0
    w2t[i] = (_Float16)W2[k*64 + c];
  }
  for (int i = tid; i < 16*64; i += blockDim.x) {
    int c = i >> 6, k = i & 63;
    float v = 0.f;
    if (c == 0) v = Wout[k*17 + 0];
    else if (c < 4) {
      float s = 0.f;
      for (int f = 0; f < 16; ++f) s += Wout[k*17 + 1 + f] * Wrgb[f*3 + (c-1)];
      v = s;
    }
    weff[i] = (_Float16)v;
  }
  if (tid < 16) {
    float v = 0.f;
    if (tid == 0) v = bout[0];
    else if (tid < 4) {
      float s = brgb[tid-1];
      for (int f = 0; f < 16; ++f) s += bout[1+f] * Wrgb[f*3 + (tid-1)];
      v = s;
    }
    beff[tid] = v;
  }
  __shared__ float red[256];
  float sum = 0.f;
  for (int j = tid; j < NRAYS; j += blockDim.x) sum += tmax[j] - tmin[j];
  red[tid] = sum;
  __syncthreads();
  for (int s = 128; s > 0; s >>= 1) {
    if (tid < s) red[tid] += red[tid + s];
    __syncthreads();
  }
  if (tid == 0) *dtp = red[0] / ((float)NRAYS * (float)SSTEPS);
}

__device__ __forceinline__ float relu0(float x) {
  return x > 0.f ? x : 0.f;
}

__device__ __forceinline__ v8f wmma_acc32(v16h a, v16h b, v8f c) {
  return __builtin_amdgcn_wmma_f32_16x16x32_f16(false, a, false, b, (short)0, c, false, false);
}

#if HAVE_WMMA_F16D
// f16-accumulate WMMA: D/C are f16 (8 halves per lane; half j -> row j+8*hi, col lane&15)
__device__ __forceinline__ v8h wmma_acc16(v16h a, v16h b, v8h c) {
  return __builtin_amdgcn_wmma_f16_16x16x32_f16(false, a, false, b, (short)0, c, false, false);
}
#endif

// B fragment 32x16 (KxN): half h of lane n holds W[K = 32*kt + 8*(n>>4) + h (h<8) / +16 (h>=8)][col]
// Wt is [col][K] f16; works for global or LDS pointers.
__device__ __forceinline__ v16h loadB(const _Float16* Wt, int c, int kt, int hi) {
  const _Float16* p = Wt + c*64 + kt*32 + hi*8;
  v8h lo = *(const v8h*)p;
  v8h hh = *(const v8h*)(p + 16);
  return __builtin_shufflevector(lo, hh, 0,1,2,3,4,5,6,7,8,9,10,11,12,13,14,15);
}

// A fragment 16x32 (MxK): lane l row m=l&15; half h -> K = 32*kt + 8*hi + h (h<8) / +16 (h>=8)
__device__ __forceinline__ v16h loadA(const _Float16* act, int m, int hi, int kt) {
  const _Float16* p = act + m*64 + kt*32 + hi*8;
  v8h lo = *(const v8h*)p;
  v8h hh = *(const v8h*)(p + 16);
  return __builtin_shufflevector(lo, hh, 0,1,2,3,4,5,6,7,8,9,10,11,12,13,14,15);
}

__device__ __forceinline__ float enc_val(float px, float py, float pz, int k) {
  if (k >= 63) return 0.f;
  if (k < 3) return (k == 0) ? px : ((k == 1) ? py : pz);
  int kk = k - 3;
  int d  = kk / 20;
  int r  = kk - d * 20;
  float p = (d == 0) ? px : ((d == 1) ? py : pz);
  int e = (r < 10) ? r : (r - 10);
  float ang = p * (float)(1u << e);
  return (r < 10) ? __sinf(ang) : __cosf(ang);
}

__global__ __launch_bounds__(256) void raymarch_kernel(
    const float* __restrict__ ox_, const float* __restrict__ oy_, const float* __restrict__ oz_,
    const float* __restrict__ dx_, const float* __restrict__ dy_, const float* __restrict__ dz_,
    const float* __restrict__ tmin_, const float* __restrict__ tmax_,
    const float* __restrict__ b1_, const float* __restrict__ b2_,
    const void* __restrict__ ws, float* __restrict__ out) {
  __shared__ _Float16 s_act[8][16*64];   // per-wave activation tile (16 samples x 64 feats)
  __shared__ float    s_ob[8][128*4];    // per-wave raw head outputs: [sample][sigma,r,g,b]

  const int wv   = threadIdx.x >> 5;
  const int lane = threadIdx.x & 31;
  const int m    = lane & 15;
  const int hi   = lane >> 4;
  _Float16* act = s_act[wv];
  float*    ob  = s_ob[wv];

  const _Float16* w1t  = (const _Float16*)ws;
  const _Float16* w2t  = (const _Float16*)((const char*)ws + 8192);
  const _Float16* wef  = (const _Float16*)((const char*)ws + 16384);
  const float*    beff = (const float*)((const char*)ws + 18432);
  const float     dtv  = *(const float*)((const char*)ws + 18496);

  // ---- hoist all weight B-fragments + biases into registers (reused every tile/ray) ----
  v16h B1f[2][4], B2f[2][4], BEf[2];
#if HAVE_WMMA_F16D
  _Float16 b1c[4], b2c[4];
#else
  float b1c[4], b2c[4];
#endif
  #pragma unroll
  for (int nt = 0; nt < 4; ++nt) {
    const int c = m + 16 * nt;
    #pragma unroll
    for (int kt = 0; kt < 2; ++kt) {
      B1f[kt][nt] = loadB(w1t, c, kt, hi);
      B2f[kt][nt] = loadB(w2t, c, kt, hi);
    }
#if HAVE_WMMA_F16D
    b1c[nt] = (_Float16)b1_[c];
    b2c[nt] = (_Float16)b2_[c];
#else
    b1c[nt] = b1_[c];
    b2c[nt] = b2_[c];
#endif
  }
  BEf[0] = loadB(wef, m, 0, hi);
  BEf[1] = loadB(wef, m, 1, hi);
  const float bec = beff[m];

  for (int ray = blockIdx.x * 8 + wv; ray < NRAYS; ray += gridDim.x * 8) {
    const float ox = ox_[ray], oy = oy_[ray], oz = oz_[ray];
    const float dx = dx_[ray], dy = dy_[ray], dz = dz_[ray];
    const float tmn = tmin_[ray], tmx = tmax_[ray];

    for (int tile = 0; tile < 8; ++tile) {
      // ---- positional encoding, generated directly in A-fragment layout ----
      const int s_idx = tile * 16 + m;
      const float t  = tmn + ((float)s_idx * (1.0f / 127.0f)) * (tmx - tmn);
      const float px = ox + dx * t, py = oy + dy * t, pz = oz + dz * t;
      v16h A0, A1;
      #pragma unroll
      for (int h = 0; h < 16; ++h) {
        const int kb = h + hi * 8 + ((h >= 8) ? 8 : 0);
        A0[h] = (_Float16)enc_val(px, py, pz, kb);
        A1[h] = (_Float16)enc_val(px, py, pz, 32 + kb);
      }

      // ---- layer 1: 16x64 = enc(16x64) @ W1(64x64), ReLU ----
      #pragma unroll
      for (int nt = 0; nt < 4; ++nt) {
        const int c = m + 16 * nt;
#if HAVE_WMMA_F16D
        const _Float16 bh = b1c[nt];
        v8h acc = {bh, bh, bh, bh, bh, bh, bh, bh};
        acc = wmma_acc16(A0, B1f[0][nt], acc);
        acc = wmma_acc16(A1, B1f[1][nt], acc);
        #pragma unroll
        for (int j = 0; j < 8; ++j) {
          _Float16 x = acc[j];
          act[(j + 8 * hi) * 64 + c] = (x > (_Float16)0.f) ? x : (_Float16)0.f;
        }
#else
        const float bc = b1c[nt];
        v8f acc = {bc, bc, bc, bc, bc, bc, bc, bc};
        acc = wmma_acc32(A0, B1f[0][nt], acc);
        acc = wmma_acc32(A1, B1f[1][nt], acc);
        #pragma unroll
        for (int j = 0; j < 8; ++j)
          act[(j + 8 * hi) * 64 + c] = (_Float16)relu0(acc[j]);
#endif
      }

      // ---- layer 2: 16x64 = h1 @ W2(64x64), ReLU ----
      v16h A20 = loadA(act, m, hi, 0);
      v16h A21 = loadA(act, m, hi, 1);
      #pragma unroll
      for (int nt = 0; nt < 4; ++nt) {
        const int c = m + 16 * nt;
#if HAVE_WMMA_F16D
        const _Float16 bh = b2c[nt];
        v8h acc = {bh, bh, bh, bh, bh, bh, bh, bh};
        acc = wmma_acc16(A20, B2f[0][nt], acc);
        acc = wmma_acc16(A21, B2f[1][nt], acc);
        #pragma unroll
        for (int j = 0; j < 8; ++j) {
          _Float16 x = acc[j];
          act[(j + 8 * hi) * 64 + c] = (x > (_Float16)0.f) ? x : (_Float16)0.f;
        }
#else
        const float bc = b2c[nt];
        v8f acc = {bc, bc, bc, bc, bc, bc, bc, bc};
        acc = wmma_acc32(A20, B2f[0][nt], acc);
        acc = wmma_acc32(A21, B2f[1][nt], acc);
        #pragma unroll
        for (int j = 0; j < 8; ++j)
          act[(j + 8 * hi) * 64 + c] = (_Float16)relu0(acc[j]);
#endif
      }

      // ---- fused output head (f32 accumulate): 16x16 = h2 @ W_eff(64x16) ----
      v16h A30 = loadA(act, m, hi, 0);
      v16h A31 = loadA(act, m, hi, 1);
      {
        v8f o = {bec, bec, bec, bec, bec, bec, bec, bec};
        o = wmma_acc32(A30, BEf[0], o);
        o = wmma_acc32(A31, BEf[1], o);
        if (m < 4) {
          #pragma unroll
          for (int r = 0; r < 8; ++r)
            ob[(tile * 16 + r + 8 * hi) * 4 + m] = o[r];
        }
      }
    }

    // ---- compositing: 4 samples per lane, wave-parallel prefix product ----
    float av[4], al[4], cr[4], cg[4], cb[4];
    #pragma unroll
    for (int q = 0; q < 4; ++q) {
      const int s2 = lane * 4 + q;
      const float sg = relu0(ob[s2 * 4 + 0]);
      const float e  = __expf(-sg * dtv);
      al[q] = 1.f - e;            // alpha
      av[q] = e + 1e-10f;         // 1 - alpha + 1e-10
      cr[q] = 1.f / (1.f + __expf(-ob[s2 * 4 + 1]));
      cg[q] = 1.f / (1.f + __expf(-ob[s2 * 4 + 2]));
      cb[q] = 1.f / (1.f + __expf(-ob[s2 * 4 + 3]));
    }
    float P = av[0] * av[1] * av[2] * av[3];
    float I = P;
    #pragma unroll
    for (int d = 1; d < 32; d <<= 1) {
      float v = __shfl_up(I, d);
      if (lane >= d) I *= v;
    }
    float E = __shfl_up(I, 1);
    if (lane == 0) E = 1.f;
    float T = E, ar = 0.f, ag = 0.f, ab = 0.f;
    #pragma unroll
    for (int q = 0; q < 4; ++q) {
      const float w = (T > 0.0001f) ? (T * al[q]) : 0.f;
      ar += w * cr[q]; ag += w * cg[q]; ab += w * cb[q];
      T *= av[q];
    }
    const float Tout = __shfl(I, 31);
    #pragma unroll
    for (int d = 16; d > 0; d >>= 1) {
      ar += __shfl_xor(ar, d);
      ag += __shfl_xor(ag, d);
      ab += __shfl_xor(ab, d);
    }
    if (lane == 0) {
      out[ray * 3 + 0] = ar;
      out[ray * 3 + 1] = ag;
      out[ray * 3 + 2] = ab;
      out[3 * NRAYS + ray] = Tout;
    }
  }
}

extern "C" void kernel_launch(void* const* d_in, const int* in_sizes, int n_in,
                              void* d_out, int out_size, void* d_ws, size_t ws_size,
                              hipStream_t stream) {
  (void)in_sizes; (void)n_in; (void)out_size; (void)ws_size;
  const float* ox   = (const float*)d_in[0];
  const float* oy   = (const float*)d_in[1];
  const float* oz   = (const float*)d_in[2];
  const float* dx   = (const float*)d_in[3];
  const float* dy   = (const float*)d_in[4];
  const float* dz   = (const float*)d_in[5];
  const float* tmin = (const float*)d_in[6];
  const float* tmax = (const float*)d_in[7];
  const float* W1   = (const float*)d_in[8];
  const float* b1   = (const float*)d_in[9];
  const float* W2   = (const float*)d_in[10];
  const float* b2   = (const float*)d_in[11];
  const float* Wout = (const float*)d_in[12];
  const float* bout = (const float*)d_in[13];
  const float* Wrgb = (const float*)d_in[14];
  const float* brgb = (const float*)d_in[15];

  prep_kernel<<<1, 256, 0, stream>>>(W1, W2, Wout, bout, Wrgb, brgb, tmin, tmax, d_ws);
  raymarch_kernel<<<2048, 256, 0, stream>>>(ox, oy, oz, dx, dy, dz, tmin, tmax,
                                            b1, b2, d_ws, (float*)d_out);
}
